// MultiHeadAttention_49177375539493
// MI455X (gfx1250) — matrix operand order repaired and checked by reading, not compiled, not run on hardware
//
#include <hip/hip_runtime.h>

#define B_SZ 4
#define S_LEN 2048
#define NH 16
#define HD 64
#define DM 1024          // QUERY_DIM == INNER
#define MROWS (B_SZ * S_LEN)

typedef __attribute__((ext_vector_type(16))) _Float16 v16h;
typedef __attribute__((ext_vector_type(8)))  _Float16 v8h;
typedef __attribute__((ext_vector_type(8)))  float    v8f;
typedef __attribute__((ext_vector_type(4)))  unsigned int u32x4;
typedef __attribute__((ext_vector_type(8)))  int i32x8;
typedef __attribute__((ext_vector_type(4)))  int i32x4;

#if __has_include(<hip/amd_detail/amd_gfx1250_TDM.h>)
#define TDM_6ARG 1
#endif

__device__ __forceinline__ v16h combine16(v8h lo, v8h hi) {
  return __builtin_shufflevector(lo, hi, 0,1,2,3,4,5,6,7,8,9,10,11,12,13,14,15);
}

// A-fragment (16x32 f16, row-major source): lane row = base row + (lane&15),
// per-lane K = (lane>>4)*8 + {0..7} and + {16..23}  -> two 16B loads.
__device__ __forceinline__ v16h load_a_frag(const _Float16* rowptr, int g) {
  const _Float16* p = rowptr + g * 8;
  return combine16(*(const v8h*)p, *(const v8h*)(p + 16));
}
// B-fragment (32x16 f16) from K-contiguous (transposed) source:
// per-lane K = (lane>>4)*8 + {0..7} and + {16..23} -> two 16B loads.
__device__ __forceinline__ v16h load_b_frag(const _Float16* rowptr, int g) {
  const _Float16* p = rowptr + g * 8;
  return combine16(*(const v8h*)p, *(const v8h*)(p + 16));
}

__device__ __forceinline__ v8f wmma_f16(v16h a, v16h b, v8f c) {
  return __builtin_amdgcn_wmma_f32_16x16x32_f16(false, a, false, b, (short)0, c,
                                                false, false);
}

// ---- Tensor Data Mover: 2-D tile (f16 elements) Global -> LDS -----------
// tile0 = row length (elements), tile1 = #rows, stride0 = row pitch (elems)
__device__ __forceinline__ void tdm_load_2d(unsigned lds_off, const void* gsrc,
                                            unsigned tile0, unsigned tile1,
                                            unsigned stride0) {
  unsigned long long ga = (unsigned long long)(uintptr_t)gsrc;
  u32x4 g0;
  g0[0] = 1u;                                      // count=1, user mode
  g0[1] = lds_off;                                 // lds_addr (bytes)
  g0[2] = (unsigned)(ga & 0xFFFFFFFFu);            // global_addr[31:0]
  g0[3] = (unsigned)((ga >> 32) & 0x01FFFFFFu) | (2u << 30);  // addr[56:32]|type=2
  i32x8 g1;
  g1[0] = 0x10000;                                 // data_size=1 (2 bytes)
  g1[1] = (int)((tile0 & 0xFFFFu) << 16);          // tensor_dim0[15:0] = tile0
  g1[2] = (int)((tile1 & 0xFFFFu) << 16);          // tensor_dim1[15:0] = tile1
  g1[3] = (int)((tile0 & 0xFFFFu) << 16);          // tile_dim0
  g1[4] = (int)(tile1 & 0xFFFFu);                  // tile_dim1 (tile_dim2=0)
  g1[5] = (int)stride0;                            // tensor_dim0_stride[31:0]
  g1[6] = 0;
  g1[7] = 0;
  i32x4 z4 = {0, 0, 0, 0};
#ifdef TDM_6ARG
  i32x8 z8 = {0, 0, 0, 0, 0, 0, 0, 0};
  __builtin_amdgcn_tensor_load_to_lds(g0, g1, z4, z4, z8, 0);
#else
  __builtin_amdgcn_tensor_load_to_lds(g0, g1, z4, z4, 0);
#endif
}

// ---------------- elementwise f32 -> f16 ----------------
__global__ void cvt_f32_f16_kernel(const float* __restrict__ in,
                                   _Float16* __restrict__ out, int n) {
  int i = blockIdx.x * blockDim.x + threadIdx.x;
  if (i < n) out[i] = (_Float16)in[i];
}

// ---------------- 1024x1024 transpose + convert: WT[n][k] = (f16)W[k][n] ----
__global__ __launch_bounds__(256)
void transpose_cvt_kernel(const float* __restrict__ W, _Float16* __restrict__ WT) {
  __shared__ float t[32][33];
  int bx = blockIdx.x * 32, by = blockIdx.y * 32;
  int tx = threadIdx.x, ty = threadIdx.y;         // 32 x 8
  for (int r = 0; r < 32; r += 8)
    t[ty + r][tx] = W[(size_t)(by + ty + r) * DM + bx + tx];
  __syncthreads();
  for (int r = 0; r < 32; r += 8)
    WT[(size_t)(bx + ty + r) * DM + by + tx] = (_Float16)t[tx][ty + r];
}

// ------- WMMA GEMM with TDM double-buffered LDS staging --------------------
// C[M,1024] = A[M,1024] * WT^T + bias
// mode 0: f16 row-major out; mode 1: f16 "V-transposed" out [B,H,64,S];
// mode 2: f32 row-major out
__global__ __launch_bounds__(256)
void gemm_wmma_kernel(const _Float16* __restrict__ A,
                      const _Float16* __restrict__ WT,
                      const float* __restrict__ bias,
                      void* __restrict__ outp, int mode) {
  const int K = DM, N = DM;
  __shared__ __align__(16) _Float16 Abuf[2][64][32];    // 2 x 4 KB
  __shared__ __align__(16) _Float16 Bbuf[2][256][32];   // 2 x 16 KB
  int w    = threadIdx.x >> 5;
  int lane = threadIdx.x & 31;
  int g    = lane >> 4;
  int lm   = lane & 15;
  int m0g  = blockIdx.x * 64;
  int n0g  = blockIdx.y * 256;
  int mrel = (w >> 2) * 32;                 // wave: 32 rows
  int nrel = (w & 3) * 64;                  // wave: 64 cols

  v8f acc[2][4];
#pragma unroll
  for (int mi = 0; mi < 2; mi++)
#pragma unroll
    for (int ni = 0; ni < 4; ni++) acc[mi][ni] = (v8f){};

  if (w == 0) {   // prologue DMA into buffer 0
    tdm_load_2d((unsigned)(uintptr_t)&Abuf[0][0][0], A + (size_t)m0g * K,
                32, 64, K);
    tdm_load_2d((unsigned)(uintptr_t)&Bbuf[0][0][0], WT + (size_t)n0g * K,
                32, 256, K);
  }

  int cur = 0;
  for (int kk = 0; kk < K; kk += 32) {
    if (w == 0) {
      if (kk + 32 < K) {   // prefetch next K-slice into the other buffer
        tdm_load_2d((unsigned)(uintptr_t)&Abuf[cur ^ 1][0][0],
                    A + (size_t)m0g * K + kk + 32, 32, 64, K);
        tdm_load_2d((unsigned)(uintptr_t)&Bbuf[cur ^ 1][0][0],
                    WT + (size_t)n0g * K + kk + 32, 32, 256, K);
        __builtin_amdgcn_s_wait_tensorcnt(2);   // current buffer complete
      } else {
        __builtin_amdgcn_s_wait_tensorcnt(0);
      }
    }
    __syncthreads();                            // LDS tiles visible to all

    v16h af[2], bf[4];
#pragma unroll
    for (int mi = 0; mi < 2; mi++) {
      const _Float16* r = &Abuf[cur][mrel + mi * 16 + lm][0];
      af[mi] = combine16(*(const v8h*)(r + g * 8), *(const v8h*)(r + g * 8 + 16));
    }
#pragma unroll
    for (int ni = 0; ni < 4; ni++) {
      const _Float16* r = &Bbuf[cur][nrel + ni * 16 + lm][0];
      bf[ni] = combine16(*(const v8h*)(r + g * 8), *(const v8h*)(r + g * 8 + 16));
    }
#pragma unroll
    for (int mi = 0; mi < 2; mi++)
#pragma unroll
      for (int ni = 0; ni < 4; ni++)
        acc[mi][ni] = wmma_f16(af[mi], bf[ni], acc[mi][ni]);

    __syncthreads();                 // reads done before buffer is overwritten
    cur ^= 1;
  }

#pragma unroll
  for (int ni = 0; ni < 4; ni++) {
    int n = n0g + nrel + ni * 16 + lm;
    float bv = bias[n];
#pragma unroll
    for (int mi = 0; mi < 2; mi++) {
      int mbase = m0g + mrel + mi * 16 + g * 8;  // 8 consecutive rows
      if (mode == 0) {
        _Float16* o = (_Float16*)outp;
#pragma unroll
        for (int i = 0; i < 8; i++)
          o[(size_t)(mbase + i) * N + n] = (_Float16)(acc[mi][ni][i] + bv);
      } else if (mode == 2) {
        float* o = (float*)outp;
#pragma unroll
        for (int i = 0; i < 8; i++)
          o[(size_t)(mbase + i) * N + n] = acc[mi][ni][i] + bv;
      } else {
        // V transposed: out[((b*NH + h)*HD + dd)*S + s], s contiguous
        _Float16* o = (_Float16*)outp;
        int b = mbase / S_LEN, s = mbase % S_LEN;
        int h = n >> 6, dd = n & 63;
        size_t base = (((size_t)b * NH + h) * HD + dd) * S_LEN + s;
        v8h pk;
#pragma unroll
        for (int i = 0; i < 8; i++) pk[i] = (_Float16)(acc[mi][ni][i] + bv);
        *(v8h*)(o + base) = pk;
      }
    }
  }
}

// ---------------- fused attention: one wave per (b, h, 16-query tile) -------
__global__ __launch_bounds__(32)
void attn_wmma_kernel(const _Float16* __restrict__ q16,  // [B,S,1024]
                      const _Float16* __restrict__ k16,  // [B,S,1024]
                      const _Float16* __restrict__ vT,   // [B,H,64,S]
                      const unsigned char* __restrict__ mask,  // [B,S,S]
                      _Float16* __restrict__ attnout) {  // [B,S,1024]
  int qt = blockIdx.x, h = blockIdx.y, b = blockIdx.z;
  int lane = threadIdx.x, g = lane >> 4, lm = lane & 15;
  int q0 = qt * 16;
  __shared__ __align__(16) _Float16 ptile[16][32];

  // Q A-fragments (two K=32 slices over d), held in registers for both passes
  const _Float16* qbase =
      q16 + ((size_t)b * S_LEN + q0 + lm) * DM + h * HD;
  v16h aq[2];
#pragma unroll
  for (int ks = 0; ks < 2; ks++) aq[ks] = load_a_frag(qbase + ks * 32, g);

  const _Float16* kbase = k16 + (size_t)b * S_LEN * DM + h * HD;
  const unsigned char* mb = mask + (size_t)b * S_LEN * S_LEN;

  // ---- pass A: row max (streaming over key tiles) ----
  float mx[8];
#pragma unroll
  for (int i = 0; i < 8; i++) mx[i] = -3.0e38f;
  for (int ct = 0; ct < S_LEN / 16; ct++) {
    int c0 = ct * 16;
    if (ct + 1 < S_LEN / 16)   // global_prefetch_b8: next key tile rows
      __builtin_prefetch(kbase + (size_t)(c0 + 16 + lm) * DM, 0, 1);
    v8f e = (v8f){};
#pragma unroll
    for (int ks = 0; ks < 2; ks++) {
      v16h bk = load_b_frag(kbase + (size_t)(c0 + lm) * DM + ks * 32, g);
      e = wmma_f16(aq[ks], bk, e);
    }
    int n = c0 + lm;
#pragma unroll
    for (int i = 0; i < 8; i++) {
      int m = q0 + g * 8 + i;
      float ev = e[i] * 0.125f;                       // 1/sqrt(64)
      if (!mb[(size_t)m * S_LEN + n]) ev = -1e9f;
      mx[i] = fmaxf(mx[i], ev);
    }
  }
#pragma unroll
  for (int i = 0; i < 8; i++) {                       // reduce over 16 lanes
    float v = mx[i];
    v = fmaxf(v, __shfl_xor(v, 1, 32));
    v = fmaxf(v, __shfl_xor(v, 2, 32));
    v = fmaxf(v, __shfl_xor(v, 4, 32));
    v = fmaxf(v, __shfl_xor(v, 8, 32));
    mx[i] = v;
  }

  // ---- pass B: exp + sum, PV accumulation via LDS-staged P tiles ----
  float sm[8];
#pragma unroll
  for (int i = 0; i < 8; i++) sm[i] = 0.f;
  v8f oacc[4];
#pragma unroll
  for (int dt = 0; dt < 4; dt++) oacc[dt] = (v8f){};
  const _Float16* vbase = vT + ((size_t)b * NH + h) * HD * S_LEN;

  for (int cb = 0; cb < S_LEN; cb += 32) {
#pragma unroll
    for (int half = 0; half < 2; half++) {
      int c0 = cb + half * 16;
      v8f e = (v8f){};
#pragma unroll
      for (int ks = 0; ks < 2; ks++) {
        v16h bk = load_b_frag(kbase + (size_t)(c0 + lm) * DM + ks * 32, g);
        e = wmma_f16(aq[ks], bk, e);
      }
      int n = c0 + lm;
#pragma unroll
      for (int i = 0; i < 8; i++) {
        int m = q0 + g * 8 + i;
        float ev = e[i] * 0.125f;
        if (!mb[(size_t)m * S_LEN + n]) ev = -1e9f;
        float p = __expf(ev - mx[i]);
        sm[i] += p;
        ptile[g * 8 + i][half * 16 + lm] = (_Float16)p;  // C-layout scatter
      }
    }
    __syncthreads();                                     // ds waits (1 wave)
    v16h ap = combine16(*(const v8h*)&ptile[lm][g * 8],
                        *(const v8h*)&ptile[lm][g * 8 + 16]);  // A-layout
#pragma unroll
    for (int dt = 0; dt < 4; dt++) {
      v16h bv = load_b_frag(vbase + (size_t)(dt * 16 + lm) * S_LEN + cb, g);
      oacc[dt] = wmma_f16(ap, bv, oacc[dt]);
    }
    __syncthreads();
  }

#pragma unroll
  for (int i = 0; i < 8; i++) {                          // row sums
    float v = sm[i];
    v += __shfl_xor(v, 1, 32);
    v += __shfl_xor(v, 2, 32);
    v += __shfl_xor(v, 4, 32);
    v += __shfl_xor(v, 8, 32);
    sm[i] = 1.0f / v;
  }
  _Float16* obase = attnout + ((size_t)b * S_LEN + q0) * DM + h * HD;
#pragma unroll
  for (int dt = 0; dt < 4; dt++)
#pragma unroll
    for (int i = 0; i < 8; i++)
      obase[(size_t)(g * 8 + i) * DM + dt * 16 + lm] =
          (_Float16)(oacc[dt][i] * sm[i]);
}

extern "C" void kernel_launch(void* const* d_in, const int* in_sizes, int n_in,
                              void* d_out, int out_size, void* d_ws,
                              size_t ws_size, hipStream_t stream) {
  const float* query   = (const float*)d_in[0];
  const float* context = (const float*)d_in[1];
  const unsigned char* mask = (const unsigned char*)d_in[2];
  const float* Wq = (const float*)d_in[3];
  const float* bq = (const float*)d_in[4];
  const float* Wk = (const float*)d_in[5];
  const float* bk = (const float*)d_in[6];
  const float* Wv = (const float*)d_in[7];
  const float* bv = (const float*)d_in[8];
  const float* Wo = (const float*)d_in[9];
  const float* bo = (const float*)d_in[10];

  char* ws = (char*)d_ws;
  size_t off = 0;
  auto take = [&](size_t bytes) {
    void* p = ws + off;
    off += (bytes + 255) & ~(size_t)255;
    return p;
  };
  const size_t actBytes = (size_t)MROWS * DM * sizeof(_Float16);  // 16 MB
  const size_t wBytes   = (size_t)DM * DM * sizeof(_Float16);     //  2 MB
  _Float16* q_in16 = (_Float16*)take(actBytes);
  _Float16* c_in16 = (_Float16*)take(actBytes);
  _Float16* WqT = (_Float16*)take(wBytes);
  _Float16* WkT = (_Float16*)take(wBytes);
  _Float16* WvT = (_Float16*)take(wBytes);
  _Float16* WoT = (_Float16*)take(wBytes);
  _Float16* q16    = (_Float16*)take(actBytes);
  _Float16* k16    = (_Float16*)take(actBytes);
  _Float16* vT16   = (_Float16*)take(actBytes);
  _Float16* attn16 = (_Float16*)take(actBytes);

  const int nAct = MROWS * DM;
  dim3 cvb(256), cvg((nAct + 255) / 256);
  cvt_f32_f16_kernel<<<cvg, cvb, 0, stream>>>(query, q_in16, nAct);
  cvt_f32_f16_kernel<<<cvg, cvb, 0, stream>>>(context, c_in16, nAct);

  dim3 tb(32, 8), tg(DM / 32, DM / 32);
  transpose_cvt_kernel<<<tg, tb, 0, stream>>>(Wq, WqT);
  transpose_cvt_kernel<<<tg, tb, 0, stream>>>(Wk, WkT);
  transpose_cvt_kernel<<<tg, tb, 0, stream>>>(Wv, WvT);
  transpose_cvt_kernel<<<tg, tb, 0, stream>>>(Wo, WoT);

  dim3 gb(256), gg(MROWS / 64, DM / 256);
  gemm_wmma_kernel<<<gg, gb, 0, stream>>>(q_in16, WqT, bq, q16, 0);
  gemm_wmma_kernel<<<gg, gb, 0, stream>>>(c_in16, WkT, bk, k16, 0);
  gemm_wmma_kernel<<<gg, gb, 0, stream>>>(c_in16, WvT, bv, vT16, 1);

  dim3 ab(32), ag(S_LEN / 16, NH, B_SZ);
  attn_wmma_kernel<<<ag, ab, 0, stream>>>(q16, k16, vT16, mask, attn16);

  gemm_wmma_kernel<<<gg, gb, 0, stream>>>(attn16, WoT, bo, d_out, 2);
}
